// Attention_layer_67877663146058
// MI455X (gfx1250) — matrix operand order repaired and checked by reading, not compiled, not run on hardware
//
#include <hip/hip_runtime.h>
#include <hip/hip_bf16.h>
#include <stdint.h>

// ---------------------------------------------------------------------------
// Attention scores + softmax for MI455X (gfx1250), bf16 WMMA path.
//   out[b,h,q,k] = softmax_k( (x@Wq^T)(x@Wk^T)^T * hd^-0.5 )
// B=8 N=1024 E=768 H=12 HD=64.  Output 402MB fp32 -> store-bound; write once.
// CDNA5 paths: v_wmma_f32_16x16x32_bf16 + global_load_async_to_lds_b128.
// ---------------------------------------------------------------------------

typedef __bf16 bf16_t;
typedef __attribute__((ext_vector_type(16))) __bf16 v16bf;
typedef __attribute__((ext_vector_type(8)))  __bf16 v8bf;
typedef __attribute__((ext_vector_type(8)))  float  v8f;

union V16 { v16bf v; v8bf h[2]; };

#define WMMA_BF16(a, b, c) \
    __builtin_amdgcn_wmma_f32_16x16x32_bf16(false, (a), false, (b), (short)0, (c), false, false)

// ---------------------------------------------------------------------------
// Kernel 1: fp32 -> bf16 convert (grid-stride)
// ---------------------------------------------------------------------------
__global__ __launch_bounds__(256) void k_convert_bf16(const float* __restrict__ src,
                                                      bf16_t* __restrict__ dst, int n) {
    for (int i = blockIdx.x * blockDim.x + threadIdx.x; i < n; i += blockDim.x * gridDim.x)
        dst[i] = (bf16_t)src[i];
}

// ---------------------------------------------------------------------------
// Kernel 2: QK projection GEMM.
//   QK[8192,1536](bf16) = Xb[8192,768](bf16) @ Wb[1536,768]^T, f32 accum.
//   Softmax scale 0.125 folded into Q columns (col < 768) at store (exact, 2^-3).
//   One wave -> 16x64 output tile; K loop 768/32 = 24 WMMAs per acc.
// ---------------------------------------------------------------------------
__global__ __launch_bounds__(256) void k_qk_gemm(const bf16_t* __restrict__ X,
                                                 const bf16_t* __restrict__ W,
                                                 bf16_t* __restrict__ QK) {
    const int lane  = threadIdx.x & 31;
    const int wave  = (blockIdx.x * blockDim.x + threadIdx.x) >> 5;
    const int mtile = wave & 511;          // 512 row tiles of 16
    const int ngrp  = wave >> 9;           // 24 col groups of 64
    const int m0 = mtile * 16;
    const int n0 = ngrp * 64;

    const int arow = m0 + (lane & 15);
    const int aoff = (lane >> 4) * 8;      // A: half-wave K split 0..7 / 8..15
    const int boff = (lane >> 4) * 8;      // B: half-wave K split 0..7 / 8..15

    v8f acc[4] = {};
    for (int k0 = 0; k0 < 768; k0 += 32) {
        V16 a;
        const bf16_t* ap = X + (size_t)arow * 768 + k0 + aoff;
        a.h[0] = *(const v8bf*)(ap);
        a.h[1] = *(const v8bf*)(ap + 16);
#pragma unroll
        for (int t = 0; t < 4; ++t) {
            const int ncol = n0 + t * 16 + (lane & 15);
            V16 b;
            const bf16_t* bp = W + (size_t)ncol * 768 + k0 + boff;
            b.h[0] = *(const v8bf*)(bp);
            b.h[1] = *(const v8bf*)(bp + 16);
            acc[t] = WMMA_BF16(a.v, b.v, acc[t]);
        }
    }
    // C/D layout: vgpr r, lanes 0-15 -> M=r, lanes 16-31 -> M=r+8; N = lane%16
    const int mrow = m0 + (lane >> 4) * 8;
#pragma unroll
    for (int t = 0; t < 4; ++t) {
        const int col = n0 + t * 16 + (lane & 15);
        const float sc = (col < 768) ? 0.125f : 1.0f;   // scale Q only
#pragma unroll
        for (int r = 0; r < 8; ++r)
            QK[(size_t)(mrow + r) * 1536 + col] = (bf16_t)(acc[t][r] * sc);
    }
}

// ---------------------------------------------------------------------------
// Kernel 3: fused scores + softmax.
//   One block = one (b,h) pair + 128-query strip. K_bh (1024x64 bf16) staged in
//   LDS via GLOBAL_LOAD_ASYNC_TO_LDS_B128 (ASYNCcnt path, no VGPR round-trip),
//   144B row stride -> conflict-free ds_load_b128 B fragments.
//   Pass 1: online per-lane max/sum over all 64 key tiles, then shfl_xor merge.
//   Pass 2: recompute tiles (bf16 WMMA, cheap) and store normalized probs once.
// ---------------------------------------------------------------------------
#define LDS_STRIDE 72   // bf16 elems per K row: 64 + 8 pad -> 144B

__global__ __launch_bounds__(256) void k_attn_softmax(const bf16_t* __restrict__ QK,
                                                      float* __restrict__ out) {
    extern __shared__ __align__(16) char smem_raw[];
    bf16_t* Ks = (bf16_t*)smem_raw;

    const int lane = threadIdx.x & 31;
    const int wid  = threadIdx.x >> 5;      // 8 waves
    const int blk    = blockIdx.x;
    const int qstrip = blk & 7;             // 8 strips of 128 rows
    const int bh     = blk >> 3;            // 0..95  (= b*12 + h)
    const int h = bh % 12;
    const int b = bh / 12;

    const bf16_t* Qbase = QK + (size_t)b * 1024 * 1536 + h * 64;         // + row*1536
    const bf16_t* Kbase = QK + (size_t)b * 1024 * 1536 + 768 + h * 64;

    // ---- async K stage: 1024 rows x 128B each, global -> LDS directly ----
    // Each lane copies 16B per issue from its global address straight to its LDS address.
    for (int r = threadIdx.x; r < 1024; r += 256) {
        const uint64_t ga = (uint64_t)(uintptr_t)(Kbase + (size_t)r * 1536);
        const uint32_t la = (uint32_t)(uintptr_t)(&Ks[r * LDS_STRIDE]);
#pragma unroll
        for (int j = 0; j < 8; ++j) {
            asm volatile("global_load_async_to_lds_b128 %0, %1, off"
                         :: "v"(la + (uint32_t)(j * 16)), "v"(ga + (uint64_t)(j * 16))
                         : "memory");
        }
    }
    asm volatile("s_wait_asynccnt 0" ::: "memory");   // own-wave async copies done
    __syncthreads();                                   // publish LDS tile to all waves

    // ---- per-wave Q fragments: 16 rows x 64 dims = 2 A tiles, kept in regs ----
    const int q0   = qstrip * 128 + wid * 16;
    const int qrow = q0 + (lane & 15);
    const int aoff = (lane >> 4) * 8;
    V16 a0, a1;
    {
        const bf16_t* qp = Qbase + (size_t)qrow * 1536;
        a0.h[0] = *(const v8bf*)(qp + 0  + aoff);
        a0.h[1] = *(const v8bf*)(qp + 16 + aoff);
        a1.h[0] = *(const v8bf*)(qp + 32 + aoff);
        a1.h[1] = *(const v8bf*)(qp + 48 + aoff);
    }

    const int boff = (lane >> 4) * 8;
    auto compute_tile = [&](int key0) -> v8f {
        const int keycol = key0 + (lane & 15);
        const bf16_t* kp = &Ks[keycol * LDS_STRIDE + boff];
        V16 b0, b1;
        b0.h[0] = *(const v8bf*)(kp);
        b0.h[1] = *(const v8bf*)(kp + 16);
        b1.h[0] = *(const v8bf*)(kp + 32);
        b1.h[1] = *(const v8bf*)(kp + 48);
        v8f acc = {};
        acc = WMMA_BF16(a0.v, b0.v, acc);
        acc = WMMA_BF16(a1.v, b1.v, acc);
        return acc;
    };

    // ---- pass 1: online max / sum-of-exp per lane (each lane: 1 col per tile) ----
    float mx[8], sm[8];
#pragma unroll
    for (int r = 0; r < 8; ++r) { mx[r] = -__builtin_inff(); sm[r] = 0.0f; }
    for (int key0 = 0; key0 < 1024; key0 += 16) {
        v8f s = compute_tile(key0);
#pragma unroll
        for (int r = 0; r < 8; ++r) {
            const float v = s[r];
            const float mn = fmaxf(mx[r], v);
            sm[r] = sm[r] * __expf(mx[r] - mn) + __expf(v - mn);
            mx[r] = mn;
        }
    }
    // merge the 16 lanes of each half-wave (row stats live across n = lane%16)
#pragma unroll
    for (int r = 0; r < 8; ++r) {
        float m = mx[r], s = sm[r];
#pragma unroll
        for (int off = 1; off < 16; off <<= 1) {
            const float mo = __shfl_xor(m, off, 32);
            const float so = __shfl_xor(s, off, 32);
            const float mn = fmaxf(m, mo);
            s = s * __expf(m - mn) + so * __expf(mo - mn);
            m = mn;
        }
        mx[r] = m;
        sm[r] = 1.0f / s;
    }

    // ---- pass 2: recompute, normalize, single store of the 402MB output ----
    const int qbase = q0 + (lane >> 4) * 8;
    const int col   = lane & 15;
    const size_t outbase = (size_t)bh * 1024 * 1024;
    for (int key0 = 0; key0 < 1024; key0 += 16) {
        v8f s = compute_tile(key0);
#pragma unroll
        for (int r = 0; r < 8; ++r) {
            const float p = __expf(s[r] - mx[r]) * sm[r];
            out[outbase + (size_t)(qbase + r) * 1024 + key0 + col] = p;
        }
    }
}

// ---------------------------------------------------------------------------
// Launch
// ---------------------------------------------------------------------------
extern "C" void kernel_launch(void* const* d_in, const int* in_sizes, int n_in,
                              void* d_out, int out_size, void* d_ws, size_t ws_size,
                              hipStream_t stream) {
    const float* x = (const float*)d_in[0];   // [8,1024,768]
    const float* W = (const float*)d_in[1];   // [2304,768] (rows 0..1535 = Q,K)
    float* out = (float*)d_out;               // [8,12,1024,1024]

    char* ws = (char*)d_ws;
    bf16_t* Xb = (bf16_t*)(ws);                         // 8192*768   = 12,582,912 B
    bf16_t* Wb = (bf16_t*)(ws + 12582912);              // 1536*768   =  2,359,296 B
    bf16_t* QK = (bf16_t*)(ws + 14942208);              // 8192*1536  = 25,165,824 B

    k_convert_bf16<<<4096, 256, 0, stream>>>(x, Xb, 8 * 1024 * 768);
    k_convert_bf16<<<2048, 256, 0, stream>>>(W, Wb, 1536 * 768);

    // 512 mtiles * 24 ngroups = 12288 waves / 8 waves per block
    k_qk_gemm<<<1536, 256, 0, stream>>>(Xb, Wb, QK);

    // 96 (b,h) pairs * 8 query strips; 144KB dynamic LDS for K_bh
    k_attn_softmax<<<768, 256, (size_t)1024 * LDS_STRIDE * sizeof(bf16_t), stream>>>(QK, out);
}
